// MultiHeadedSelfAttention_91336774517471
// MI455X (gfx1250) — compile-verified, matrix operand order checked by reading, not run on hardware
//
#include <hip/hip_runtime.h>
#include <hip/hip_bf16.h>

typedef __bf16 bf16_t;
typedef __attribute__((ext_vector_type(16))) __bf16 v16bf;
typedef __attribute__((ext_vector_type(8)))  __bf16 v8bf;
typedef __attribute__((ext_vector_type(4)))  __bf16 v4bf;
typedef __attribute__((ext_vector_type(8)))  float  v8f;
typedef __attribute__((ext_vector_type(4)))  float  v4f;
typedef __attribute__((ext_vector_type(4)))  unsigned int v4u;
typedef __attribute__((ext_vector_type(8)))  int v8i;
typedef __attribute__((ext_vector_type(4)))  int v4i;

constexpr int kB = 2;
constexpr int kS = 2048;
constexpr int kD = 1024;
constexpr int kH = 16;
constexpr int SSTR = 2052;    // score panel row stride (pad vs 64 LDS banks)
constexpr int XSTR = 36;      // LDS x-panel row stride (32 + 4 pad, 16B aligned)

// Pack 16 f32 -> v16bf via packed converts (v_cvt_pk_bf16_f32), no b16 movs.
__device__ __forceinline__ v16bf pack16(v4f a, v4f b, v4f c, v4f d) {
    v4bf x0 = __builtin_convertvector(a, v4bf);
    v4bf x1 = __builtin_convertvector(b, v4bf);
    v4bf x2 = __builtin_convertvector(c, v4bf);
    v4bf x3 = __builtin_convertvector(d, v4bf);
    v8bf lo = __builtin_shufflevector(x0, x1, 0, 1, 2, 3, 4, 5, 6, 7);
    v8bf hi = __builtin_shufflevector(x2, x3, 0, 1, 2, 3, 4, 5, 6, 7);
    return __builtin_shufflevector(lo, hi, 0, 1, 2, 3, 4, 5, 6, 7,
                                   8, 9, 10, 11, 12, 13, 14, 15);
}

#if defined(__has_builtin)
#if __has_builtin(__builtin_amdgcn_tensor_load_to_lds) && __has_builtin(__builtin_amdgcn_s_wait_tensorcnt)
#define HAVE_TDM 1
#endif
#endif

#ifdef HAVE_TDM
// Issue one 2D TENSOR_LOAD_TO_LDS with a hand-packed D# (groups 0/1, 2D tile).
// dims/strides in elements (data_size = 4 bytes). Executed by one wave.
// 6-arg builtin form: (v4u g0, v8i g1, v4i g2, v4i g3, v8i g4, i32 cpol).
__device__ __forceinline__ void tdm_load_2d(unsigned lds_off, const void* gptr,
                                            unsigned tensor_d0, unsigned tensor_d1,
                                            unsigned tile_d0,   unsigned tile_d1,
                                            unsigned stride0,
                                            unsigned pad_en, unsigned pad_int,
                                            unsigned pad_amt)
{
    unsigned long long ga = (unsigned long long)gptr;
    v4u g0;
    g0[0] = 1u;                                          // count=1, user D#
    g0[1] = lds_off;                                     // lds_addr
    g0[2] = (unsigned)(ga & 0xffffffffu);                // global_addr[31:0]
    g0[3] = (unsigned)((ga >> 32) & 0x1ffffffu)          // global_addr[56:32]
            | 0x80000000u;                               // type=2 (image)
    v8i g1;
    g1[0] = (int)((2u << 16) | (pad_en << 20) | (pad_int << 22) | (pad_amt << 25));
    g1[1] = (int)((tensor_d0 & 0xffffu) << 16);          // tensor_dim0[15:0]
    g1[2] = (int)(((tensor_d0 >> 16) & 0xffffu) | ((tensor_d1 & 0xffffu) << 16));
    g1[3] = (int)(((tensor_d1 >> 16) & 0xffffu) | ((tile_d0 & 0xffffu) << 16));
    g1[4] = (int)(tile_d1 & 0xffffu);                    // tile_dim1, tile_dim2=0
    g1[5] = (int)stride0;                                // tensor_dim0_stride[31:0]
    g1[6] = 0;
    g1[7] = 0;
    v4i z4 = {0, 0, 0, 0};
    v8i z8 = {0, 0, 0, 0, 0, 0, 0, 0};
    __builtin_amdgcn_tensor_load_to_lds(g0, g1, z4, z4, z8, 0);
}
__device__ __forceinline__ unsigned lds_offset_of(const void* p) {
    return (unsigned)(unsigned long long)(uintptr_t)p;   // LDS aperture: addr[31:0]
}
#endif

// ---------------------------------------------------------------------------
// Kernel 1: Q/K/V projection, blocked 128x64 per 256-thread block.
// x/W panels staged in ping-pong LDS buffers via TDM (tensor_load_to_lds):
// the next panel's DMA runs concurrently with the current panel's WMMAs
// (s_wait_tensorcnt 2 = wait only for the older descriptor pair).
// Results stored pre-swizzled as WMMA fragment blobs:
//   Qblob[bh][qt][chunk][lane][16]   (A-fragments for scores)
//   Kblob[bh][kt][chunk][lane][16]   (B-fragments for scores)
//   Vblob[bh][nt][kc][lane][16]      (B-fragments for attn @ V)
// ---------------------------------------------------------------------------
__global__ __launch_bounds__(256) void qkv_proj_kernel(
    const float* __restrict__ x,
    const float* __restrict__ Wq, const float* __restrict__ bq,
    const float* __restrict__ Wk, const float* __restrict__ bk,
    const float* __restrict__ Wv, const float* __restrict__ bv,
    bf16_t* __restrict__ Qblob, bf16_t* __restrict__ Kblob,
    bf16_t* __restrict__ Vblob)
{
    __shared__ float xs[2][128 * XSTR];   // 2 x (128 x 32 f32, stride 36)
    __shared__ float wsh[2][32 * 64];     // 2 x (32 x 64 f32)

    const int lane   = threadIdx.x & 31;
    const int wave   = threadIdx.x >> 5;  // 0..7
    const int lane15 = lane & 15;
    const int koff   = (lane < 16) ? 0 : 8;

    const int mat = blockIdx.x >> 9;                 // 0=Q 1=K 2=V
    const int rem = blockIdx.x & 511;
    const int m0  = (rem >> 4) << 7;                 // 32 M tiles * 128
    const int n0  = (rem & 15) << 6;                 // 16 N tiles * 64

    const float* Wm   = (mat == 0) ? Wq : ((mat == 1) ? Wk : Wv);
    const float* bias = (mat == 0) ? bq : ((mat == 1) ? bk : bv);

    v8f acc[4] = {v8f{}, v8f{}, v8f{}, v8f{}};

#ifdef HAVE_TDM
    if (wave == 0) {                                  // prologue: panel 0
        tdm_load_2d(lds_offset_of(xs[0]), x + (size_t)m0 * kD,
                    kD, kB * kS, 32, 128, kD, 1, 4, 3);
        tdm_load_2d(lds_offset_of(wsh[0]), Wm + n0,
                    kD, kD, 64, 32, kD, 0, 0, 0);
    }
#endif

    for (int it = 0; it < kD / 32; ++it) {
        const int buf = it & 1;
#ifdef HAVE_TDM
        if (wave == 0) {
            if (it + 1 < kD / 32) {
                const int k1 = (it + 1) * 32;
                // issue next panel into the other buffer (overlaps compute)
                tdm_load_2d(lds_offset_of(xs[buf ^ 1]), x + (size_t)m0 * kD + k1,
                            kD, kB * kS, 32, 128, kD, 1, 4, 3);
                tdm_load_2d(lds_offset_of(wsh[buf ^ 1]), Wm + (size_t)k1 * kD + n0,
                            kD, kD, 64, 32, kD, 0, 0, 0);
                __builtin_amdgcn_s_wait_tensorcnt(2);  // older pair complete
            } else {
                __builtin_amdgcn_s_wait_tensorcnt(0);
            }
        }
#else
        {
            const int k0 = it * 32;
            for (int e = threadIdx.x; e < 128 * 8; e += 256) {   // v4f chunks
                const int r = e >> 3, c = (e & 7) << 2;
                *(v4f*)(xs[buf] + r * XSTR + c) =
                    *(const v4f*)(x + (size_t)(m0 + r) * kD + k0 + c);
            }
            for (int e = threadIdx.x; e < 32 * 16; e += 256) {
                const int r = e >> 4, c = (e & 15) << 2;
                *(v4f*)(wsh[buf] + r * 64 + c) =
                    *(const v4f*)(Wm + (size_t)(k0 + r) * kD + n0 + c);
            }
        }
#endif
        __syncthreads();                              // panel `buf` ready

        // A fragment from LDS (wave's 16-row strip), packed converts
        const float* ar = xs[buf] + (wave * 16 + lane15) * XSTR + koff;
        const v16bf A = pack16(*(const v4f*)(ar),      *(const v4f*)(ar + 4),
                               *(const v4f*)(ar + 16), *(const v4f*)(ar + 20));

        // Materialize ALL four B fragments first, then issue the four
        // independent WMMAs back-to-back.
        v16bf Bfs[4];
        #pragma unroll
        for (int nt = 0; nt < 4; ++nt) {
            const float* br = wsh[buf] + koff * 64 + nt * 16 + lane15;
            v4f b0 = {br[0 * 64],  br[1 * 64],  br[2 * 64],  br[3 * 64]};
            v4f b1 = {br[4 * 64],  br[5 * 64],  br[6 * 64],  br[7 * 64]};
            v4f b2 = {br[16 * 64], br[17 * 64], br[18 * 64], br[19 * 64]};
            v4f b3 = {br[20 * 64], br[21 * 64], br[22 * 64], br[23 * 64]};
            Bfs[nt] = pack16(b0, b1, b2, b3);
        }
        #pragma unroll
        for (int nt = 0; nt < 4; ++nt) {
            acc[nt] = __builtin_amdgcn_wmma_f32_16x16x32_bf16(
                          false, A, false, Bfs[nt], (short)0, acc[nt], false, false);
        }
        __syncthreads();                              // reads done; buf reusable
    }

    // Store pre-swizzled fragment blobs (scattered b16 stores, done once)
    #pragma unroll
    for (int nt = 0; nt < 4; ++nt) {
        const int f     = n0 + nt * 16 + lane15;
        const float bc  = bias[f];
        const int h     = f >> 6;
        const int kk    = f & 31;
        const int chunk = (f >> 5) & 1;
        const int loffF = (kk & 8) ? 16 : 0;
        const int jF    = (kk & 7) + ((kk & 16) >> 1);
        #pragma unroll
        for (int r = 0; r < 8; ++r) {
            const int mloc = r + ((lane >> 4) << 3);
            const int t    = m0 + wave * 16 + mloc;            // token index
            const bf16_t v = (bf16_t)(acc[nt][r] + bc);
            const int b = t >> 11, s = t & 2047;
            if (mat == 0) {
                const int qt = s >> 4, m = s & 15;
                Qblob[((size_t)((b * kH + h) * 128 + qt) * 2 + chunk) * 512
                      + (m + loffF) * 16 + jF] = v;
            } else if (mat == 1) {
                const int kt = s >> 4, n = s & 15;
                Kblob[((size_t)((b * kH + h) * 128 + kt) * 2 + chunk) * 512
                      + (n + loffF) * 16 + jF] = v;
            } else {
                const int w = f & 63;
                const int ntv = w >> 4, n = w & 15;
                const int kci = s >> 5, ks = s & 31;
                const int loffV = (ks & 8) ? 16 : 0;
                const int jV    = (ks & 7) + ((ks & 16) >> 1);
                Vblob[((size_t)((b * kH + h) * 4 + ntv) * 64 + kci) * 512
                      + (n + loffV) * 16 + jV] = v;
            }
        }
    }
}

// ---------------------------------------------------------------------------
// Kernel 2: fused scores -> mask -> softmax -> attn write -> attn @ V.
// One block = 16 queries x one (b,h); 4 waves; 16x2048 f32 score panel in LDS.
// All Q/K/V fragment loads are single aligned 32B blob loads per lane.
// ---------------------------------------------------------------------------
__global__ __launch_bounds__(128) void attn_kernel(
    const float* __restrict__ mask,
    const bf16_t* __restrict__ Qblob, const bf16_t* __restrict__ Kblob,
    const bf16_t* __restrict__ Vblob,
    float* __restrict__ hout, float* __restrict__ attn_out)
{
    extern __shared__ float sc[];                    // [16][SSTR]
    const int lane = threadIdx.x & 31;
    const int wave = threadIdx.x >> 5;
    const int blk  = blockIdx.x;
    const int qt   = blk & 127;
    const int bh   = blk >> 7;                       // 0..31
    const int b    = bh >> 4;
    const int h    = bh & 15;
    const int q0   = qt * 16;
    const int koff = (lane < 16) ? 0 : 8;

    // Q A-fragments: one 32B load per chunk per lane
    const bf16_t* qbase = Qblob + ((size_t)(bh * 128 + qt) * 2) * 512 + lane * 16;
    const v16bf A0 = *(const v16bf*)(qbase);
    const v16bf A1 = *(const v16bf*)(qbase + 512);

    // ---- scores: waves stripe the 128 key tiles -------------------------
    #pragma unroll 2
    for (int kt = wave; kt < kS / 16; kt += 4) {
        const bf16_t* kbase = Kblob + ((size_t)(bh * 128 + kt) * 2) * 512 + lane * 16;
        const v16bf B0 = *(const v16bf*)(kbase);
        const v16bf B1 = *(const v16bf*)(kbase + 512);
        v8f c = {};
        c = __builtin_amdgcn_wmma_f32_16x16x32_bf16(false, A0, false, B0,
                                                    (short)0, c, false, false);
        c = __builtin_amdgcn_wmma_f32_16x16x32_bf16(false, A1, false, B1,
                                                    (short)0, c, false, false);
        const int key     = kt * 16 + (lane & 15);
        const float mbias = -10000.0f * (1.0f - mask[b * kS + key]);
        #pragma unroll
        for (int r = 0; r < 8; ++r) {
            const int m = r + ((lane >> 4) << 3);
            sc[m * SSTR + key] = c[r] * 0.125f + mbias;      // 1/sqrt(64)
        }
    }
    __syncthreads();

    // ---- softmax: wave w owns rows 4w..4w+3, lanes stripe the keys ------
    float* attn_bh = attn_out + (size_t)bh * kS * kS + (size_t)q0 * kS;
    #pragma unroll
    for (int r = 0; r < 4; ++r) {
        const int m = wave * 4 + r;
        float* rowp = sc + m * SSTR;
        float mx = -3.0e38f;
        for (int k = lane; k < kS; k += 32) mx = fmaxf(mx, rowp[k]);
        #pragma unroll
        for (int off = 16; off > 0; off >>= 1) mx = fmaxf(mx, __shfl_xor(mx, off, 32));
        float sum = 0.0f;
        for (int k = lane; k < kS; k += 32) {
            const float e = __expf(rowp[k] - mx);
            rowp[k] = e;
            sum += e;
        }
        #pragma unroll
        for (int off = 16; off > 0; off >>= 1) sum += __shfl_xor(sum, off, 32);
        const float inv = 1.0f / sum;
        float* arow = attn_bh + (size_t)m * kS;
        for (int k = lane; k < kS; k += 32) {        // coalesced attn write
            const float a = rowp[k] * inv;
            rowp[k] = a;
            arow[k] = a;
        }
    }
    __syncthreads();

    // ---- h = attn @ V: wave w owns w-slice [16w, 16w+16) ----------------
    // Unrolled x2 with two independent accumulators: breaks the serial
    // C-accumulation chain between consecutive WMMAs.
    const bf16_t* vbase = Vblob + ((size_t)(bh * 4 + wave) * 64) * 512 + lane * 16;
    v8f acc0 = {}, acc1 = {};
    const int am = lane & 15;
    for (int kc = 0; kc < kS; kc += 64) {
        const float* ap = sc + am * SSTR + kc + koff;        // 16B-aligned
        const v16bf Af0 = pack16(*(const v4f*)(ap),      *(const v4f*)(ap + 4),
                                 *(const v4f*)(ap + 16), *(const v4f*)(ap + 20));
        const v16bf Af1 = pack16(*(const v4f*)(ap + 32), *(const v4f*)(ap + 36),
                                 *(const v4f*)(ap + 48), *(const v4f*)(ap + 52));
        const v16bf Bf0 = *(const v16bf*)(vbase + (size_t)(kc >> 5) * 512);
        const v16bf Bf1 = *(const v16bf*)(vbase + (size_t)((kc >> 5) + 1) * 512);
        acc0 = __builtin_amdgcn_wmma_f32_16x16x32_bf16(false, Af0, false, Bf0,
                                                       (short)0, acc0, false, false);
        acc1 = __builtin_amdgcn_wmma_f32_16x16x32_bf16(false, Af1, false, Bf1,
                                                       (short)0, acc1, false, false);
    }
    #pragma unroll
    for (int r = 0; r < 8; ++r) {
        const int m = r + ((lane >> 4) << 3);
        hout[(size_t)(b * kS + q0 + m) * kD + h * 64 + wave * 16 + (lane & 15)]
            = acc0[r] + acc1[r];
    }
}

extern "C" void kernel_launch(void* const* d_in, const int* in_sizes, int n_in,
                              void* d_out, int out_size, void* d_ws, size_t ws_size,
                              hipStream_t stream) {
    const float* x    = (const float*)d_in[0];
    const float* mask = (const float*)d_in[1];
    const float* Wq   = (const float*)d_in[2];
    const float* bq   = (const float*)d_in[3];
    const float* Wk   = (const float*)d_in[4];
    const float* bk   = (const float*)d_in[5];
    const float* Wv   = (const float*)d_in[6];
    const float* bv   = (const float*)d_in[7];

    float* hout = (float*)d_out;                               // B*S*D
    float* attn = hout + (size_t)kB * kS * kD;                 // B*H*S*S

    bf16_t* Qblob = (bf16_t*)d_ws;                             // 8 MB
    bf16_t* Kblob = Qblob + (size_t)4194304;                   // 8 MB
    bf16_t* Vblob = Kblob + (size_t)4194304;                   // 8 MB

    // 3 mats * (32 M-tiles x 16 N-tiles) blocks, 256 threads each
    qkv_proj_kernel<<<1536, 256, 0, stream>>>(x, Wq, bq, Wk, bk, Wv, bv,
                                              Qblob, Kblob, Vblob);
    // one block per (b, h, 16-query tile); 16*SSTR floats of dynamic LDS
    attn_kernel<<<kB * kH * (kS / 16), 128, 16 * SSTR * sizeof(float), stream>>>(
        mask, Qblob, Kblob, Vblob, hout, attn);
}
